// MultiHeadAttention_38122129719779
// MI455X (gfx1250) — compile-verified, matrix operand order checked by reading, not run on hardware
//
#include <hip/hip_runtime.h>

// ---------------------------------------------------------------------------
// MHA for gfx1250 (CDNA5, wave32): bf16 WMMA GEMMs + flash attention.
// cvt/transpose -> QKV GEMMs (Q pre-scaled by 1/sqrt(DH)) -> V transpose ->
// flash attention -> output GEMM (+bias, f32 out).
// GEMM uses double-buffered LDS fed by GLOBAL_LOAD_ASYNC_TO_LDS_B128.
// ---------------------------------------------------------------------------

typedef __bf16 bf16_t;
typedef __attribute__((ext_vector_type(16))) __bf16 v16bf;
typedef __attribute__((ext_vector_type(8)))  __bf16 v8bf;
typedef __attribute__((ext_vector_type(8)))  float  v8f;

static constexpr int Bb   = 2;
static constexpr int Ss   = 2048;
static constexpr int DIN  = 1024;
static constexpr int DOUT = 1024;
static constexpr int Hh   = 16;
static constexpr int DHd  = 64;

#if defined(__has_builtin)
#if __has_builtin(__builtin_amdgcn_global_load_async_to_lds_b128)
#define HAVE_ASYNC_LDS 1
#endif
#if __has_builtin(__builtin_amdgcn_s_wait_asynccnt)
#define HAVE_WAIT_ASYNC 1
#endif
#endif

__device__ __forceinline__ bf16_t f2bf(float f) {
  unsigned u = __builtin_bit_cast(unsigned, f);
  unsigned r = (u + 0x7FFFu + ((u >> 16) & 1u)) >> 16;
  return __builtin_bit_cast(bf16_t, (unsigned short)r);
}

__device__ __forceinline__ v8f wmma_bf16(v16bf a, v16bf b, v8f c) {
  return __builtin_amdgcn_wmma_f32_16x16x32_bf16(
      false, a, false, b, (short)0, c, false, false);
}

// 16-byte global -> LDS copy; async on CDNA5 (ASYNCcnt), sync fallback.
__device__ __forceinline__ void cp16_g2l(const bf16_t* g, bf16_t* l) {
#if defined(HAVE_ASYNC_LDS)
  typedef int v4i __attribute__((vector_size(16)));
  typedef __attribute__((address_space(1))) v4i* gp_t;   // global int4*
  typedef __attribute__((address_space(3))) v4i* lp_t;   // LDS int4*
  __builtin_amdgcn_global_load_async_to_lds_b128((gp_t)g, (lp_t)l, 0, 0);
#else
  *reinterpret_cast<v8bf*>(l) = *reinterpret_cast<const v8bf*>(g);
#endif
}

__device__ __forceinline__ void wait_async() {
#if defined(HAVE_WAIT_ASYNC)
  __builtin_amdgcn_s_wait_asynccnt(0);
#elif defined(HAVE_ASYNC_LDS)
  asm volatile("s_wait_asynccnt 0" ::: "memory");
#endif
}

// 16x32 bf16 fragment load (A-layout; also B stored N-major, K-contiguous).
// ISA 7.12.2: lane l -> row (l&15); K chunks [(l>>4)*8 ..] and [.. +16].
__device__ __forceinline__ v16bf load_frag(const bf16_t* base, int stride) {
  int lane = threadIdx.x & 31;
  int row  = lane & 15;
  int kb   = (lane >> 4) * 8;
  union { v16bf v; v8bf h[2]; } u;
  const bf16_t* p = base + (size_t)row * stride + kb;
  u.h[0] = *reinterpret_cast<const v8bf*>(p);
  u.h[1] = *reinterpret_cast<const v8bf*>(p + 16);
  return u.v;
}

__device__ __forceinline__ float rowmax16(float v) {
#pragma unroll
  for (int m = 1; m < 16; m <<= 1) v = fmaxf(v, __shfl_xor(v, m, 32));
  return v;
}
__device__ __forceinline__ float rowsum16(float v) {
#pragma unroll
  for (int m = 1; m < 16; m <<= 1) v += __shfl_xor(v, m, 32);
  return v;
}

__device__ __forceinline__ void store_val(float* p, float v)  { *p = v; }
__device__ __forceinline__ void store_val(bf16_t* p, float v) { *p = f2bf(v); }

// ------------------------------- converters --------------------------------

__global__ void __launch_bounds__(256) cvt_kernel(const float* __restrict__ s,
                                                  bf16_t* __restrict__ d, int n) {
  int i = blockIdx.x * 256 + threadIdx.x;
  if (i < n) d[i] = f2bf(s[i]);
}

// Wt[n][k] = W[k][n]  (1024 x 1024, f32 -> bf16)
__global__ void __launch_bounds__(256) wtrans_kernel(const float* __restrict__ s,
                                                     bf16_t* __restrict__ d) {
  int i = blockIdx.x * 256 + threadIdx.x;   // i = n*1024 + k
  int n = i >> 10, k = i & 1023;
  d[i] = f2bf(s[k * 1024 + n]);
}

// Vt[b][h][d][s] = V[b][s][h*64+d]
__global__ void __launch_bounds__(256) vtrans_kernel(const bf16_t* __restrict__ v,
                                                     bf16_t* __restrict__ vt) {
  int i = blockIdx.x * 256 + threadIdx.x;
  int s = i & (Ss - 1);
  int r = i >> 11;
  int d = r & (DHd - 1);  r >>= 6;
  int h = r & (Hh - 1);
  int b = r >> 4;
  vt[i] = v[((size_t)(b * Ss + s)) * DOUT + h * DHd + d];
}

// ------------------------------- WMMA GEMM ---------------------------------
// C[M,N] = oscale * (A[M,K] x Bt[N,K]^T) + bias.  Block 256 thr (8 waves),
// tile 128x128, K-step 32, double-buffered LDS via async global->LDS loads.
template <typename OUT>
__global__ void __launch_bounds__(256)
gemm_bf16(const bf16_t* __restrict__ A, const bf16_t* __restrict__ Bt,
          OUT* __restrict__ C, const float* __restrict__ bias, float oscale,
          int M, int N, int K) {
  __shared__ bf16_t As[2][128][40];
  __shared__ bf16_t Bs[2][128][40];

  const int tid  = threadIdx.x;
  const int lane = tid & 31;
  const int wave = tid >> 5;
  const int wm = wave >> 1;          // 0..3
  const int wn = wave & 1;           // 0..1
  const int bm = blockIdx.y * 128;
  const int bn = blockIdx.x * 128;

  v8f acc[2][4];
#pragma unroll
  for (int mt = 0; mt < 2; ++mt)
#pragma unroll
    for (int nt = 0; nt < 4; ++nt) acc[mt][nt] = {};

  const int lrow = tid >> 1;
  const int lcol = (tid & 1) * 16;

  auto issue = [&](int buf, int k0) {
    const bf16_t* ag = A + (size_t)(bm + lrow) * K + k0 + lcol;
    cp16_g2l(ag,     &As[buf][lrow][lcol]);
    cp16_g2l(ag + 8, &As[buf][lrow][lcol + 8]);
    const bf16_t* bg = Bt + (size_t)(bn + lrow) * K + k0 + lcol;
    cp16_g2l(bg,     &Bs[buf][lrow][lcol]);
    cp16_g2l(bg + 8, &Bs[buf][lrow][lcol + 8]);
  };

  issue(0, 0);
  int buf = 0;
  for (int k0 = 0; k0 < K; k0 += 32, buf ^= 1) {
    wait_async();          // this wave's async loads for `buf` are in LDS
    __syncthreads();       // everyone's loads landed; prior reads retired
    if (k0 + 32 < K) issue(buf ^ 1, k0 + 32);   // overlap with compute

    v16bf af[2], bff[4];
#pragma unroll
    for (int mt = 0; mt < 2; ++mt)
      af[mt] = load_frag(&As[buf][wm * 32 + mt * 16][0], 40);
#pragma unroll
    for (int nt = 0; nt < 4; ++nt)
      bff[nt] = load_frag(&Bs[buf][wn * 64 + nt * 16][0], 40);
#pragma unroll
    for (int mt = 0; mt < 2; ++mt)
#pragma unroll
      for (int nt = 0; nt < 4; ++nt)
        acc[mt][nt] = wmma_bf16(af[mt], bff[nt], acc[mt][nt]);
    __syncthreads();
  }

  const int n = lane & 15, half = lane >> 4;
#pragma unroll
  for (int mt = 0; mt < 2; ++mt)
#pragma unroll
    for (int nt = 0; nt < 4; ++nt) {
      int col = bn + wn * 64 + nt * 16 + n;
      float bv = bias ? bias[col] : 0.0f;
#pragma unroll
      for (int r = 0; r < 8; ++r) {
        int row = bm + wm * 32 + mt * 16 + r + 8 * half;
        store_val(&C[(size_t)row * N + col], acc[mt][nt][r] * oscale + bv);
      }
    }
}

// ---------------------------- flash attention ------------------------------
// grid (S/64, B*H), block 128 (4 waves); each wave owns 16 query rows.
// Q is pre-scaled by 1/sqrt(DH).  Q,K: [B,S,H,DH].  Vt: [B,H,DH,S].
__global__ void __launch_bounds__(128)
attn_kernel(const bf16_t* __restrict__ Q, const bf16_t* __restrict__ Kb,
            const bf16_t* __restrict__ Vt, bf16_t* __restrict__ ctx) {
  __shared__ bf16_t Pbuf[4][16][32];

  const int lane = threadIdx.x & 31;
  const int wave = threadIdx.x >> 5;
  const int bh = blockIdx.y;
  const int b  = bh >> 4;
  const int h  = bh & 15;
  const int q0 = blockIdx.x * 64 + wave * 16;
  const int n = lane & 15, half = lane >> 4;

  const bf16_t* qbase = Q + (size_t)(b * Ss + q0) * DOUT + h * DHd;
  const bf16_t* khead = Kb + (size_t)(b * Ss) * DOUT + h * DHd;
  const bf16_t* vhead = Vt + (size_t)((b * Hh + h) * DHd) * Ss;

  v16bf qa0 = load_frag(qbase, DOUT);
  v16bf qa1 = load_frag(qbase + 32, DOUT);

  float m[8], l[8];
  v8f accO[4];
#pragma unroll
  for (int r = 0; r < 8; ++r) { m[r] = -1e30f; l[r] = 0.0f; }
#pragma unroll
  for (int nt = 0; nt < 4; ++nt) accO[nt] = {};

  const int q_hi = q0 + 15;

  auto block = [&](int j, bool masked) {
    // prefetch next key block (one 128B line covers a full DH row)
    __builtin_prefetch(khead + (size_t)(j + 32 + n) * DOUT, 0, 1);
    __builtin_prefetch(khead + (size_t)(j + 48 + n) * DOUT, 0, 1);

    v8f sc[2];
#pragma unroll
    for (int t = 0; t < 2; ++t) {
      const bf16_t* kbase = khead + (size_t)(j + t * 16) * DOUT;
      v16bf kf0 = load_frag(kbase, DOUT);
      v16bf kf1 = load_frag(kbase + 32, DOUT);
      v8f c = {};
      c = wmma_bf16(qa0, kf0, c);
      c = wmma_bf16(qa1, kf1, c);
      sc[t] = c;
    }

#pragma unroll
    for (int r = 0; r < 8; ++r) {
      float s0 = sc[0][r];
      float s1 = sc[1][r];
      if (masked) {
        const int qg = q0 + r + 8 * half;
        if (j + n > qg)      s0 = -1e30f;
        if (j + 16 + n > qg) s1 = -1e30f;
      }
      float mn = fmaxf(m[r], rowmax16(fmaxf(s0, s1)));
      float alpha = __expf(m[r] - mn);
      float p0 = __expf(s0 - mn);
      float p1 = __expf(s1 - mn);
      l[r] = l[r] * alpha + rowsum16(p0 + p1);
      m[r] = mn;
#pragma unroll
      for (int nt = 0; nt < 4; ++nt) accO[nt][r] *= alpha;
      Pbuf[wave][r + 8 * half][n]      = f2bf(p0);
      Pbuf[wave][r + 8 * half][16 + n] = f2bf(p1);
    }

    v16bf pa = load_frag(&Pbuf[wave][0][0], 32);
#pragma unroll
    for (int nt = 0; nt < 4; ++nt) {
      v16bf vb = load_frag(vhead + (size_t)(nt * 16) * Ss + j, Ss);
      accO[nt] = wmma_bf16(pa, vb, accO[nt]);
    }
  };

  int j = 0;
  for (; j + 31 <= q0; j += 32) block(j, false);   // bulk: no mask VALU
  for (; j <= q_hi;    j += 32) block(j, true);    // 1-2 diagonal blocks

#pragma unroll
  for (int nt = 0; nt < 4; ++nt)
#pragma unroll
    for (int r = 0; r < 8; ++r) {
      float o = accO[nt][r] / l[r];
      size_t idx = (size_t)(b * Ss + q0 + r + 8 * half) * DOUT +
                   h * DHd + nt * 16 + n;
      ctx[idx] = f2bf(o);
    }
}

// ------------------------------- launcher ----------------------------------

extern "C" void kernel_launch(void* const* d_in, const int* in_sizes, int n_in,
                              void* d_out, int out_size, void* d_ws, size_t ws_size,
                              hipStream_t stream) {
  (void)in_sizes; (void)n_in; (void)out_size; (void)ws_size;
  const float* x  = (const float*)d_in[0];
  const float* Wq = (const float*)d_in[1];
  const float* Wk = (const float*)d_in[2];
  const float* Wv = (const float*)d_in[3];
  const float* Wo = (const float*)d_in[4];
  const float* bo = (const float*)d_in[5];
  float* out = (float*)d_out;

  const size_t NX = (size_t)Bb * Ss * DIN;
  const size_t NW = (size_t)DIN * DOUT;

  bf16_t* w   = (bf16_t*)d_ws;
  bf16_t* xb  = w; w += NX;
  bf16_t* wqt = w; w += NW;
  bf16_t* wkt = w; w += NW;
  bf16_t* wvt = w; w += NW;
  bf16_t* wot = w; w += NW;
  bf16_t* Qb  = w; w += NX;
  bf16_t* Kbb = w; w += NX;
  bf16_t* Vb  = w; w += NX;
  bf16_t* Vt  = w; w += NX;
  bf16_t* ctx = xb;                    // xb dead after QKV GEMMs

  cvt_kernel<<<(int)(NX / 256), 256, 0, stream>>>(x, xb, (int)NX);
  wtrans_kernel<<<(int)(NW / 256), 256, 0, stream>>>(Wq, wqt);
  wtrans_kernel<<<(int)(NW / 256), 256, 0, stream>>>(Wk, wkt);
  wtrans_kernel<<<(int)(NW / 256), 256, 0, stream>>>(Wv, wvt);
  wtrans_kernel<<<(int)(NW / 256), 256, 0, stream>>>(Wo, wot);

  dim3 gg(DOUT / 128, (Bb * Ss) / 128);
  const float qscale = 0.125f;         // 1/sqrt(64), folded into Q
  gemm_bf16<bf16_t><<<gg, 256, 0, stream>>>(xb, wqt, Qb,  nullptr, qscale, Bb * Ss, DOUT, DIN);
  gemm_bf16<bf16_t><<<gg, 256, 0, stream>>>(xb, wkt, Kbb, nullptr, 1.0f,   Bb * Ss, DOUT, DIN);
  gemm_bf16<bf16_t><<<gg, 256, 0, stream>>>(xb, wvt, Vb,  nullptr, 1.0f,   Bb * Ss, DOUT, DIN);

  vtrans_kernel<<<(int)(NX / 256), 256, 0, stream>>>(Vb, Vt);

  attn_kernel<<<dim3(Ss / 64, Bb * Hh), 128, 0, stream>>>(Qb, Kbb, Vt, ctx);

  gemm_bf16<float><<<gg, 256, 0, stream>>>(ctx, wot, out, bo, 1.0f, Bb * Ss, DOUT, DOUT);
}
